// TemporalAttention_55808805044790
// MI455X (gfx1250) — matrix operand order repaired and checked by reading, not compiled, not run on hardware
//
#include <hip/hip_runtime.h>
#include <hip/hip_bf16.h>

// ---------------------------------------------------------------------------
// CDNA5 (gfx1250) bf16 WMMA implementation of TemporalAttention.
// All heavy math via v_wmma_f32_16x16x32_bf16 (f32 accumulate); GEMM staging
// uses global_load_async_to_lds_b128 (ASYNCcnt) when the toolchain has it.
// ---------------------------------------------------------------------------

typedef __bf16 bf16;
typedef __attribute__((ext_vector_type(8)))  __bf16 v8bf;
typedef __attribute__((ext_vector_type(16))) __bf16 v16bf;
typedef __attribute__((ext_vector_type(8)))  float  v8f;

#ifndef __has_builtin
#define __has_builtin(x) 0
#endif

#if __has_builtin(__builtin_amdgcn_global_load_async_to_lds_b128)
#define USE_ASYNC_LDS 1
#else
#define USE_ASYNC_LDS 0
#endif

__device__ __forceinline__ bf16 f2bf(float f) {
  union { float f; unsigned u; } a; a.f = f;
  unsigned r = a.u + 0x7FFFu + ((a.u >> 16) & 1u);   // round-to-nearest-even
  union { unsigned short s; __bf16 b; } o; o.s = (unsigned short)(r >> 16);
  return o.b;
}

__device__ __forceinline__ v16bf make_v16(v8bf lo, v8bf hi) {
  v16bf r;
#pragma unroll
  for (int i = 0; i < 8; ++i) { r[i] = lo[i]; r[8 + i] = hi[i]; }
  return r;
}

__device__ __forceinline__ v8f wmma_bf16(v16bf a, v16bf b, v8f c) {
  // D = A(16x32 bf16) x B(32x16 bf16) + C(16x16 f32)
  return __builtin_amdgcn_wmma_f32_16x16x32_bf16(false, a, false, b,
                                                 (short)0, c, false, false);
}

#if USE_ASYNC_LDS
// Builtin parameter types (from clang diagnostic): v4i in addrspace(1) / (3).
typedef int v4i_vs __attribute__((vector_size(4 * sizeof(int))));
typedef __attribute__((address_space(1))) v4i_vs* as1_v4i;
typedef __attribute__((address_space(3))) v4i_vs* as3_v4i;

__device__ __forceinline__ void async_b128(const bf16* g, bf16* l) {
  void* gp = (void*)g;
  void* lp = (void*)l;
  __builtin_amdgcn_global_load_async_to_lds_b128((as1_v4i)gp, (as3_v4i)lp, 0, 0);
}
__device__ __forceinline__ void wait_async0() {
#if __has_builtin(__builtin_amdgcn_s_wait_asynccnt)
  __builtin_amdgcn_s_wait_asynccnt(0);
#else
  asm volatile("s_wait_asynccnt 0x0" ::: "memory");
#endif
}
#endif

// ---------------------------------------------------------------------------
// Generic GEMM: C[M,N] = alpha * (A[M,K](bf16) * W[N,K](bf16)^T + bias[N] (+pos))
// Tile 128x128, BK=32, 256 threads (8 waves). Each wave: 32x64 sub-tile ->
// 8 WMMA per k-step. Async global->LDS double buffering: one barrier per step.
// ---------------------------------------------------------------------------
__global__ __launch_bounds__(256)
void gemm_bf16_kernel(const bf16* __restrict__ A, const bf16* __restrict__ W,
                      const float* __restrict__ bias,
                      const float* __restrict__ pos, int posS,
                      const float* __restrict__ alpha_ptr, int alpha_idx,
                      float* __restrict__ Cf, bf16* __restrict__ Cb,
                      int M, int N, int K, int accumulate)
{
  __shared__ bf16 As[2][128 * 40];   // 128 rows x 32 cols, padded to 40 (80B rows)
  __shared__ bf16 Ws[2][128 * 40];
  const int t    = threadIdx.x;
  const int wave = t >> 5, lane = t & 31;
  const int lh   = lane >> 4, ll = lane & 15;
  const int wr   = wave >> 1;          // 0..3 -> 32-row stripes
  const int wc   = wave & 1;           // 0..1 -> 64-col stripes
  const int m0   = blockIdx.y * 128;
  const int n0   = blockIdx.x * 128;
  const int lrow = t >> 1;             // 0..127
  const int lk   = (t & 1) * 16;       // 0,16

  v8f acc[2][4];
#pragma unroll
  for (int fa = 0; fa < 2; ++fa)
#pragma unroll
    for (int fb = 0; fb < 4; ++fb) acc[fa][fb] = (v8f){};

  const int nsteps = K >> 5;   // K/32

#if USE_ASYNC_LDS
  // -------- async double-buffered staging (1 barrier per k-step) --------
  {
    const bf16* ga = &A[(size_t)(m0 + lrow) * K + lk];
    const bf16* gw = &W[(size_t)(n0 + lrow) * K + lk];
    async_b128(ga,     &As[0][lrow * 40 + lk]);
    async_b128(ga + 8, &As[0][lrow * 40 + lk + 8]);
    async_b128(gw,     &Ws[0][lrow * 40 + lk]);
    async_b128(gw + 8, &Ws[0][lrow * 40 + lk + 8]);
  }
  for (int s = 0; s < nsteps; ++s) {
    wait_async0();
    __syncthreads();             // tile s visible to all waves; buf s^1 free
    if (s + 1 < nsteps) {
      const int kk = (s + 1) << 5;
      const int nb = (s + 1) & 1;
      const bf16* ga = &A[(size_t)(m0 + lrow) * K + kk + lk];
      const bf16* gw = &W[(size_t)(n0 + lrow) * K + kk + lk];
      async_b128(ga,     &As[nb][lrow * 40 + lk]);
      async_b128(ga + 8, &As[nb][lrow * 40 + lk + 8]);
      async_b128(gw,     &Ws[nb][lrow * 40 + lk]);
      async_b128(gw + 8, &Ws[nb][lrow * 40 + lk + 8]);
    }
    const int buf = s & 1;
#else
  // -------- fallback: load -> regs -> ds_store, single buffer ----------
  for (int s = 0; s < nsteps; ++s) {
    const int kk = s << 5;
    v8bf ra0 = *(const v8bf*)&A[(size_t)(m0 + lrow) * K + kk + lk];
    v8bf ra1 = *(const v8bf*)&A[(size_t)(m0 + lrow) * K + kk + lk + 8];
    v8bf rw0 = *(const v8bf*)&W[(size_t)(n0 + lrow) * K + kk + lk];
    v8bf rw1 = *(const v8bf*)&W[(size_t)(n0 + lrow) * K + kk + lk + 8];
    *(v8bf*)&As[0][lrow * 40 + lk]     = ra0;
    *(v8bf*)&As[0][lrow * 40 + lk + 8] = ra1;
    *(v8bf*)&Ws[0][lrow * 40 + lk]     = rw0;
    *(v8bf*)&Ws[0][lrow * 40 + lk + 8] = rw1;
    __syncthreads();
    const int buf = 0;
#endif
    // -------- fragments + 8 WMMA per wave --------
    const int ar0 = wr * 32 + ll;
    v16bf af0 = make_v16(*(const v8bf*)&As[buf][ar0 * 40 + lh * 8],
                         *(const v8bf*)&As[buf][ar0 * 40 + 16 + lh * 8]);
    v16bf af1 = make_v16(*(const v8bf*)&As[buf][(ar0 + 16) * 40 + lh * 8],
                         *(const v8bf*)&As[buf][(ar0 + 16) * 40 + 16 + lh * 8]);
    v16bf bfr[4];
#pragma unroll
    for (int fb = 0; fb < 4; ++fb) {
      const int br = wc * 64 + fb * 16 + ll;
      bfr[fb] = make_v16(*(const v8bf*)&Ws[buf][br * 40 + lh * 8],
                         *(const v8bf*)&Ws[buf][br * 40 + 16 + lh * 8]);
    }
#pragma unroll
    for (int fb = 0; fb < 4; ++fb) {
      acc[0][fb] = wmma_bf16(af0, bfr[fb], acc[0][fb]);
      acc[1][fb] = wmma_bf16(af1, bfr[fb], acc[1][fb]);
    }
#if !USE_ASYNC_LDS
    __syncthreads();
#endif
  }

  const float alpha = alpha_ptr ? alpha_ptr[alpha_idx] : 1.0f;
#pragma unroll
  for (int fa = 0; fa < 2; ++fa) {
#pragma unroll
    for (int fb = 0; fb < 4; ++fb) {
      const int col = n0 + wc * 64 + fb * 16 + ll;
#pragma unroll
      for (int j = 0; j < 8; ++j) {
        const int m = m0 + wr * 32 + fa * 16 + lh * 8 + j;
        float v = acc[fa][fb][j] + bias[col];
        if (posS > 0) v += pos[(size_t)(m % posS) * N + col];
        v *= alpha;
        const size_t idx = (size_t)m * N + col;
        if (Cf) { if (accumulate) Cf[idx] += v; else Cf[idx] = v; }
        if (Cb) Cb[idx] = f2bf(v);
      }
    }
  }
}

// ---------------------------------------------------------------------------
// Flash attention (transposed form), band masked OUT (PyTorch bool-mask).
// One wave per 16-query tile; S^T = K*Q^T (2 frags x 2 WMMA), o^T = V^T*P^T
// (4 WMMA). Online softmax lives per-lane (col = query = lane&15).
// ---------------------------------------------------------------------------
__global__ __launch_bounds__(128)
void attn_kernel(const bf16* __restrict__ qkv, const bf16* __restrict__ vt,
                 const unsigned char* __restrict__ mask,
                 bf16* __restrict__ obuf, int scale, int S, int H, int nh)
{
  const int dh   = 64;
  const int ld   = 3 * H;
  const int wave = threadIdx.x >> 5, lane = threadIdx.x & 31;
  const int lh   = lane >> 4, ll = lane & 15;
  const int qt   = blockIdx.x * 4 + wave;        // 16-query tile index
  const int bh   = blockIdx.y;                   // b*nh + head
  const int b    = bh / nh, hd = bh % nh;
  const int q0   = qt * 16;
  const int qa   = q0 + ll;                      // this lane's query (column)

  const bf16* qb = qkv + (size_t)b * S * ld + hd * dh;          // Q
  const bf16* kb = qkv + (size_t)b * S * ld + H + hd * dh;      // K
  const bf16* vb = vt + (size_t)bh * dh * S;                    // V^T [64][S]

  // Q^T B-fragments (col = query, K-dim = dh contiguous)
  const bf16* qrow = qb + (size_t)(q0 + ll) * ld;
  v16bf bq0 = make_v16(*(const v8bf*)(qrow + lh * 8),
                       *(const v8bf*)(qrow + 16 + lh * 8));
  v16bf bq1 = make_v16(*(const v8bf*)(qrow + 32 + lh * 8),
                       *(const v8bf*)(qrow + 48 + lh * 8));

  float mrun = -1e30f, lrun = 0.0f;
  v8f o0 = {}, o1 = {}, o2 = {}, o3 = {};

  for (int k0 = 0; k0 < S; k0 += 32) {
    // K A-fragments: rows = keys, K-dim = dh
    const bf16* kr0 = kb + (size_t)(k0 + ll) * ld;
    const bf16* kr1 = kb + (size_t)(k0 + 16 + ll) * ld;
    v16bf ka00 = make_v16(*(const v8bf*)(kr0 + lh * 8),      *(const v8bf*)(kr0 + 16 + lh * 8));
    v16bf ka01 = make_v16(*(const v8bf*)(kr0 + 32 + lh * 8), *(const v8bf*)(kr0 + 48 + lh * 8));
    v16bf ka10 = make_v16(*(const v8bf*)(kr1 + lh * 8),      *(const v8bf*)(kr1 + 16 + lh * 8));
    v16bf ka11 = make_v16(*(const v8bf*)(kr1 + 32 + lh * 8), *(const v8bf*)(kr1 + 48 + lh * 8));

    v8f z = {};
    v8f s0 = wmma_bf16(ka01, bq1, wmma_bf16(ka00, bq0, z));   // keys k0..k0+15
    v8f s1 = wmma_bf16(ka11, bq1, wmma_bf16(ka10, bq0, z));   // keys k0+16..k0+31

    // scale + mask (band is masked OUT; band width <= 64, so only tiles
    // overlapping [q0-scale, q0+15] can be masked)
    float p0[8], p1[8];
    float tm = -1e30f;
    const bool nearband = (k0 + 31 >= q0 - scale) && (k0 <= q0 + 15);
#pragma unroll
    for (int j = 0; j < 8; ++j) {
      const int ka_ = k0 + lh * 8 + j;
      const int kb_ = ka_ + 16;
      float v0 = s0[j] * 0.125f;   // 1/sqrt(64)
      float v1 = s1[j] * 0.125f;
      if (nearband) {
        if (ka_ <= qa && ka_ + scale >= qa && mask[(size_t)qa * S + ka_]) v0 = -1e30f;
        if (kb_ <= qa && kb_ + scale >= qa && mask[(size_t)qa * S + kb_]) v1 = -1e30f;
      }
      p0[j] = v0; p1[j] = v1;
      tm = fmaxf(tm, fmaxf(v0, v1));
    }
    tm = fmaxf(tm, __shfl_xor(tm, 16, 32));     // combine both key-halves
    const float mnew = fmaxf(mrun, tm);
    const float corr = __expf(mrun - mnew);
    lrun *= corr;
#pragma unroll
    for (int j = 0; j < 8; ++j) { o0[j] *= corr; o1[j] *= corr; o2[j] *= corr; o3[j] *= corr; }
    float rs = 0.0f;
#pragma unroll
    for (int j = 0; j < 8; ++j) {
      p0[j] = __expf(p0[j] - mnew);
      p1[j] = __expf(p1[j] - mnew);
      rs += p0[j] + p1[j];
    }
    rs += __shfl_xor(rs, 16, 32);
    lrun += rs;
    mrun = mnew;

    // Build P^T B-fragment (K-dim = 32 keys, col = query) via one cross-half
    // shuffle: lanes 0-15 need keys 0..15 (frag s0), lanes 16-31 keys 16..31.
    v16bf pb;
#pragma unroll
    for (int j = 0; j < 8; ++j) {
      const float t0 = __shfl_xor(p0[j], 16, 32);
      const float t1 = __shfl_xor(p1[j], 16, 32);
      pb[j]     = f2bf(lh ? t1 : p0[j]);
      pb[8 + j] = f2bf(lh ? p1[j] : t0);
    }

    // V^T A-fragments: rows = dh, K-dim = keys (contiguous in vt)
    const bf16* vr = vb + (size_t)ll * S + k0 + lh * 16;
    v16bf va0 = make_v16(*(const v8bf*)(vr),              *(const v8bf*)(vr + 8));
    v16bf va1 = make_v16(*(const v8bf*)(vr + 16 * S),     *(const v8bf*)(vr + 16 * S + 8));
    v16bf va2 = make_v16(*(const v8bf*)(vr + 32 * S),     *(const v8bf*)(vr + 32 * S + 8));
    v16bf va3 = make_v16(*(const v8bf*)(vr + 48 * S),     *(const v8bf*)(vr + 48 * S + 8));
    o0 = wmma_bf16(va0, pb, o0);
    o1 = wmma_bf16(va1, pb, o1);
    o2 = wmma_bf16(va2, pb, o2);
    o3 = wmma_bf16(va3, pb, o3);
  }

  const float inv = 1.0f / lrun;
  const int s_ = q0 + ll;
  const size_t base = ((size_t)b * S + s_) * H + hd * dh;
#pragma unroll
  for (int j = 0; j < 8; ++j) {
    obuf[base + 0  + lh * 8 + j] = f2bf(o0[j] * inv);
    obuf[base + 16 + lh * 8 + j] = f2bf(o1[j] * inv);
    obuf[base + 32 + lh * 8 + j] = f2bf(o2[j] * inv);
    obuf[base + 48 + lh * 8 + j] = f2bf(o3[j] * inv);
  }
}

// ---------------------------------------------------------------------------
// Small helper kernels
// ---------------------------------------------------------------------------
__global__ void cvt_kernel(const float* __restrict__ src, bf16* __restrict__ dst, int n) {
  int i = blockIdx.x * blockDim.x + threadIdx.x;
  if (i < n) dst[i] = f2bf(src[i]);
}

// vt[bh][d][s] = V part of qkv
__global__ void vtrans_kernel(const bf16* __restrict__ qkv, bf16* __restrict__ vt,
                              int S, int H, int nh, int total) {
  int i = blockIdx.x * blockDim.x + threadIdx.x;
  if (i >= total) return;
  const int s  = i % S;
  const int r  = i / S;
  const int d  = r % 64;
  const int bh = r / 64;
  const int b  = bh / nh, hd = bh % nh;
  vt[i] = qkv[((size_t)b * S + s) * (3 * H) + 2 * H + hd * 64 + d];
}

__global__ void softmax3_kernel(const float* __restrict__ sw, float* __restrict__ w) {
  if (threadIdx.x == 0) {
    const float a = sw[0], b = sw[1], c = sw[2];
    const float m = fmaxf(a, fmaxf(b, c));
    const float ea = __expf(a - m), eb = __expf(b - m), ec = __expf(c - m);
    const float s = ea + eb + ec;
    w[0] = ea / s; w[1] = eb / s; w[2] = ec / s;
  }
}

// res = out + h; LayerNorm over H=512; one block per row.
__global__ __launch_bounds__(256)
void ln_kernel(const float* __restrict__ o, const float* __restrict__ h,
               const float* __restrict__ gamma, const float* __restrict__ beta,
               float* __restrict__ y, int H)
{
  __shared__ float s1[256], s2[256];
  const int m = blockIdx.x, t = threadIdx.x;
  const size_t base = (size_t)m * H;
  const float a  = o[base + t]       + h[base + t];
  const float bb = o[base + 256 + t] + h[base + 256 + t];
  s1[t] = a + bb;
  s2[t] = a * a + bb * bb;
  __syncthreads();
  for (int st = 128; st > 0; st >>= 1) {
    if (t < st) { s1[t] += s1[t + st]; s2[t] += s2[t + st]; }
    __syncthreads();
  }
  const float mu   = s1[0] / (float)H;
  const float var  = s2[0] / (float)H - mu * mu;
  const float invs = rsqrtf(var + 1e-5f);
  y[base + t]       = (a  - mu) * invs * gamma[t]       + beta[t];
  y[base + 256 + t] = (bb - mu) * invs * gamma[256 + t] + beta[256 + t];
}

// ---------------------------------------------------------------------------
// Host orchestration
// ---------------------------------------------------------------------------
extern "C" void kernel_launch(void* const* d_in, const int* in_sizes, int n_in,
                              void* d_out, int out_size, void* d_ws, size_t ws_size,
                              hipStream_t stream)
{
  (void)in_sizes; (void)n_in; (void)out_size; (void)ws_size;
  const int B = 2, S = 2048, Din = 256, H = 512, nh = 8;
  const int M = B * S;

  const float*         x    = (const float*)d_in[0];
  const unsigned char* mask = (const unsigned char*)d_in[1];   // bool (1 byte)
  const float*         pos  = (const float*)d_in[2];
  const float*         W_in = (const float*)d_in[3];
  const float*         b_in = (const float*)d_in[4];
  const float*         inw  = (const float*)d_in[5];
  const float*         inb  = (const float*)d_in[6];
  const float*         outw = (const float*)d_in[7];
  const float*         outb = (const float*)d_in[8];
  const float*         sw   = (const float*)d_in[9];
  const float*         Wout = (const float*)d_in[10];
  const float*         bout = (const float*)d_in[11];
  const float*         lng  = (const float*)d_in[12];
  const float*         lnb  = (const float*)d_in[13];

  char*  ws  = (char*)d_ws;
  size_t off = 0;
  auto alloc = [&](size_t bytes) -> char* {
    char* p = ws + off;
    off += (bytes + 255) & ~(size_t)255;
    return p;
  };
  float* wsm     = (float*)alloc(16);                         // softmax(scale_weights)
  bf16*  x16     = (bf16*)alloc((size_t)M * Din * 2);
  bf16*  wIn16   = (bf16*)alloc((size_t)H * Din * 2);
  bf16*  inw16   = (bf16*)alloc((size_t)3 * 3 * H * H * 2);
  bf16*  outw16  = (bf16*)alloc((size_t)3 * H * H * 2);
  bf16*  wOut16  = (bf16*)alloc((size_t)H * H * 2);
  float* hf      = (float*)alloc((size_t)M * H * 4);
  bf16*  h16     = (bf16*)alloc((size_t)M * H * 2);
  bf16*  qkv16   = (bf16*)alloc((size_t)M * 3 * H * 2);
  bf16*  vt16    = (bf16*)alloc((size_t)B * nh * 64 * S * 2);
  bf16*  o16     = (bf16*)alloc((size_t)M * H * 2);
  float* combf   = (float*)alloc((size_t)M * H * 4);
  bf16*  comb16  = (bf16*)alloc((size_t)M * H * 2);
  float* outf    = (float*)alloc((size_t)M * H * 4);

  auto cvt = [&](const float* s, bf16* d, int n) {
    cvt_kernel<<<(n + 255) / 256, 256, 0, stream>>>(s, d, n);
  };

  // 0) precision conversion of activations/weights to bf16
  cvt(x,    x16,    M * Din);
  cvt(W_in, wIn16,  H * Din);
  cvt(inw,  inw16,  3 * 3 * H * H);
  cvt(outw, outw16, 3 * H * H);
  cvt(Wout, wOut16, H * H);
  softmax3_kernel<<<1, 32, 0, stream>>>(sw, wsm);

  // 1) h = x @ W_in^T + b_in + pos  (keep f32 for residual, bf16 for GEMMs)
  gemm_bf16_kernel<<<dim3(H / 128, M / 128), 256, 0, stream>>>(
      x16, wIn16, b_in, pos, S, nullptr, 0, hf, h16, M, H, Din, 0);

  const int scales[3] = {5, 21, 63};
  for (int i = 0; i < 3; ++i) {
    // 2) qkv = h @ in_proj_w[i]^T + in_proj_b[i]   (bf16 out)
    gemm_bf16_kernel<<<dim3(3 * H / 128, M / 128), 256, 0, stream>>>(
        h16, inw16 + (size_t)i * 3 * H * H, inb + (size_t)i * 3 * H,
        nullptr, 0, nullptr, 0, nullptr, qkv16, M, 3 * H, H, 0);

    // 3) V transpose for contiguous V^T A-fragments
    const int vtn = B * nh * 64 * S;
    vtrans_kernel<<<(vtn + 255) / 256, 256, 0, stream>>>(qkv16, vt16, S, H, nh, vtn);

    // 4) flash attention (band masked out)
    attn_kernel<<<dim3(S / 64, B * nh), 128, 0, stream>>>(
        qkv16, vt16, mask, o16, scales[i], S, H, nh);

    // 5) combined += w[i] * (o @ out_proj_w[i]^T + out_proj_b[i])
    gemm_bf16_kernel<<<dim3(H / 128, M / 128), 256, 0, stream>>>(
        o16, outw16 + (size_t)i * H * H, outb + (size_t)i * H,
        nullptr, 0, wsm, i, combf, nullptr, M, H, H, i > 0 ? 1 : 0);
  }

  // 6) out = combined @ W_out^T + b_out
  cvt(combf, comb16, M * H);
  gemm_bf16_kernel<<<dim3(H / 128, M / 128), 256, 0, stream>>>(
      comb16, wOut16, bout, nullptr, 0, nullptr, 0, outf, nullptr, M, H, H, 0);

  // 7) res = out + h; LayerNorm -> d_out (f32)
  ln_kernel<<<M, 256, 0, stream>>>(outf, hf, lng, lnb, (float*)d_out, H);
}
